// ReconBlock_15968688407225
// MI455X (gfx1250) — matrix operand order repaired and checked by reading, not compiled, not run on hardware
//
#include <hip/hip_runtime.h>

typedef __attribute__((ext_vector_type(16))) _Float16 v16h;
typedef __attribute__((ext_vector_type(8)))  float    v8f;

#define CCH 32
#define NMTX 18                    // 3 axes * 3 taps * 2 column-halves
#define B_DWORDS (NMTX * 32 * 8)   // packed f16 B fragments, 8 dwords per lane

// single-instruction packed f32->f16 convert: v_cvt_pk_rtz_f16_f32
__device__ __forceinline__ unsigned pkh(float lo, float hi) {
    auto p = __builtin_amdgcn_cvt_pkrtz(lo, hi);   // __fp16 ext_vector(2)
    return __builtin_bit_cast(unsigned, p);
}

union V16U8 { unsigned u[8]; uint4 q[2]; v16h v; };

// A-fragment (16x32 f16, CDNA5 layout): lane holds row M = lane&15.
// lanes 0-15:  K = 0..7 (VGPR0-3) and 16..23 (VGPR4-7)
// lanes 16-31: K = 8..15           and 24..31
// Loads are ALWAYS issued (caller clamps row to a valid one); missing rows are
// zeroed by masking the packed dwords -> no EXEC divergence, loads clause up.
__device__ __forceinline__ v16h load_A(const float* __restrict__ row, int chBase, bool ok) {
    float4 a0 = *(const float4*)(row + chBase + 0);
    float4 a1 = *(const float4*)(row + chBase + 4);
    float4 a2 = *(const float4*)(row + chBase + 16);
    float4 a3 = *(const float4*)(row + chBase + 20);
    unsigned msk = ok ? 0xFFFFFFFFu : 0u;
    V16U8 r;
    r.u[0] = pkh(a0.x, a0.y) & msk; r.u[1] = pkh(a0.z, a0.w) & msk;
    r.u[2] = pkh(a1.x, a1.y) & msk; r.u[3] = pkh(a1.z, a1.w) & msk;
    r.u[4] = pkh(a2.x, a2.y) & msk; r.u[5] = pkh(a2.z, a2.w) & msk;
    r.u[6] = pkh(a3.x, a3.y) & msk; r.u[7] = pkh(a3.z, a3.w) & msk;
    return r.v;
}

// B-fragment: pre-packed by prep kernel, 8 dwords (16 f16) per lane.
__device__ __forceinline__ v16h load_B(const uint4* __restrict__ Bp, int mtx, int lane) {
    V16U8 r;
    const uint4* p = Bp + ((mtx * 32 + lane) << 1);
    r.q[0] = p[0];
    r.q[1] = p[1];
    return r.v;
}

// One-block prep: pack W[3][3][32][32] f32 into f16 CDNA5 B layout in ws,
// and fold BN into per-channel scale/shift.
__global__ __launch_bounds__(256) void recon_prep(
        const float* __restrict__ W, const float* __restrict__ gamma,
        const float* __restrict__ beta, const float* __restrict__ mean,
        const float* __restrict__ var, float* __restrict__ ws) {
    unsigned* Bp = (unsigned*)ws;
    int t = threadIdx.x;
    for (int d = t; d < B_DWORDS; d += 256) {
        int m    = d >> 8;           // matrix index 0..17
        int lane = (d >> 3) & 31;    // destination lane
        int j    = d & 7;            // VGPR within fragment
        int at   = m >> 1;           // axis*3 + tap
        int h    = m & 1;            // output-column half
        int n    = (lane & 15) + 16 * h;             // output channel (column)
        int k    = ((lane >> 4) << 3) + 2 * j + ((j >> 2) << 3);   // input channel (K)
        float w0 = W[(at * CCH + k    ) * CCH + n];
        float w1 = W[(at * CCH + k + 1) * CCH + n];
        Bp[d] = pkh(w0, w1);
    }
    float* S = ws + B_DWORDS;
    float* T = S + 96;
    for (int i = t; i < 96; i += 256) {
        float s = gamma[i] * __frsqrt_rn(var[i] + 1e-5f);
        S[i] = s;
        T[i] = beta[i] - mean[i] * s;
    }
}

__device__ __forceinline__ float sigmoidf_fast(float x) {
    return 1.0f / (1.0f + __expf(-x));
}

__global__ __launch_bounds__(256) void recon_fused(
        const float* __restrict__ F, const int* __restrict__ nprev,
        const int* __restrict__ nnext, const float* __restrict__ ws,
        float* __restrict__ out, int N) {
    const uint4* Bp = (const uint4*)ws;
    const float* S  = ws + B_DWORDS;
    const float* T  = S + 96;

    int lane = threadIdx.x & 31;
    int wave = threadIdx.x >> 5;
    long tile = (long)blockIdx.x * 8 + wave;
    long r0 = tile * 16;
    if (r0 >= N) return;                 // wave-uniform: EXEC stays all-1s

    int m      = lane & 15;              // row within tile this lane feeds (A side)
    int hi     = lane >> 4;
    int chBase = hi * 8;                 // A-layout channel base for this lane half
    long r     = r0 + m;
    bool rok   = r < (long)N;
    long rr    = rok ? r : 0;            // clamped: always a valid row

    v16h Af = load_A(F + rr * CCH, chBase, rok);

    int n0 = m;                          // D-layout: lane -> column
    int n1 = m + 16;

    v8f acc0 = {0.f,0.f,0.f,0.f,0.f,0.f,0.f,0.f};
    v8f acc1 = acc0;

    #pragma unroll
    for (int a = 0; a < 3; ++a) {
        // unconditional index loads from clamped row; mask applied afterwards
        int pidx = nprev[a * N + rr];
        int nidx = nnext[a * N + rr];
        bool pok = rok && (pidx >= 0);
        bool nok = rok && (nidx >= 0);
        long pr = pok ? (long)pidx : 0;  // clamped: always a valid row
        long nr = nok ? (long)nidx : 0;
        v16h Ap = load_A(F + pr * CCH, chBase, pok);
        v16h An = load_A(F + nr * CCH, chBase, nok);

        v8f c0 = {0.f,0.f,0.f,0.f,0.f,0.f,0.f,0.f};
        v8f c1 = c0;

        {   // tap 0: previous neighbor
            v16h b0 = load_B(Bp, (a * 3 + 0) * 2 + 0, lane);
            v16h b1 = load_B(Bp, (a * 3 + 0) * 2 + 1, lane);
            c0 = __builtin_amdgcn_wmma_f32_16x16x32_f16(false, Ap, false, b0, (short)0, c0, false, false);
            c1 = __builtin_amdgcn_wmma_f32_16x16x32_f16(false, Ap, false, b1, (short)0, c1, false, false);
        }
        {   // tap 1: center
            v16h b0 = load_B(Bp, (a * 3 + 1) * 2 + 0, lane);
            v16h b1 = load_B(Bp, (a * 3 + 1) * 2 + 1, lane);
            c0 = __builtin_amdgcn_wmma_f32_16x16x32_f16(false, Af, false, b0, (short)0, c0, false, false);
            c1 = __builtin_amdgcn_wmma_f32_16x16x32_f16(false, Af, false, b1, (short)0, c1, false, false);
        }
        {   // tap 2: next neighbor
            v16h b0 = load_B(Bp, (a * 3 + 2) * 2 + 0, lane);
            v16h b1 = load_B(Bp, (a * 3 + 2) * 2 + 1, lane);
            c0 = __builtin_amdgcn_wmma_f32_16x16x32_f16(false, An, false, b0, (short)0, c0, false, false);
            c1 = __builtin_amdgcn_wmma_f32_16x16x32_f16(false, An, false, b1, (short)0, c1, false, false);
        }

        // BN (folded) + sigmoid, accumulate across axes. Per lane the column is
        // fixed, so scale/shift are lane scalars.
        float s0 = S[a * CCH + n0], t0 = T[a * CCH + n0];
        float s1 = S[a * CCH + n1], t1 = T[a * CCH + n1];
        #pragma unroll
        for (int j = 0; j < 8; ++j) {
            acc0[j] += sigmoidf_fast(c0[j] * s0 + t0);
            acc1[j] += sigmoidf_fast(c1[j] * s1 + t1);
        }
    }

    // Gate with input features and store. D layout: lane half selects row block.
    #pragma unroll
    for (int j = 0; j < 8; ++j) {
        long row = r0 + j + 8 * hi;
        if (row < (long)N) {
            out[row * CCH + n0] = acc0[j] * F[row * CCH + n0];
            out[row * CCH + n1] = acc1[j] * F[row * CCH + n1];
        }
    }
}

extern "C" void kernel_launch(void* const* d_in, const int* in_sizes, int n_in,
                              void* d_out, int out_size, void* d_ws, size_t ws_size,
                              hipStream_t stream) {
    const float* F     = (const float*)d_in[0];
    const int*   nprev = (const int*)d_in[1];
    const int*   nnext = (const int*)d_in[2];
    const float* W     = (const float*)d_in[3];
    const float* gamma = (const float*)d_in[4];
    const float* beta  = (const float*)d_in[5];
    const float* mean  = (const float*)d_in[6];
    const float* var   = (const float*)d_in[7];
    int N = in_sizes[0] / CCH;

    recon_prep<<<1, 256, 0, stream>>>(W, gamma, beta, mean, var, (float*)d_ws);

    int tiles  = (N + 15) / 16;
    int blocks = (tiles + 7) / 8;   // 8 waves (tiles) per 256-thread block
    recon_fused<<<blocks, 256, 0, stream>>>(F, nprev, nnext, (const float*)d_ws,
                                            (float*)d_out, N);
}
